// ContrastiveLoss_6674379178306
// MI455X (gfx1250) — compile-verified, matrix operand order checked by reading, not run on hardware
//
#include <hip/hip_runtime.h>

// ---------------------------------------------------------------------------
// Focal contrastive loss, fused for MI455X (gfx1250, wave32, WMMA + TDM).
// logits (8192x8192) never touch memory: only per-row / per-column exp-sums
// (fixed shift 1/T, valid since |dot|<=1) and the f32 diagonal are kept.
// B-matrix tiles are staged into LDS by the Tensor Data Mover (double
// buffered, s_wait_tensorcnt) and shared by all 8 waves of a workgroup,
// cutting L2 traffic 8x vs per-wave streaming.
// ---------------------------------------------------------------------------

typedef _Float16 half2v __attribute__((ext_vector_type(2)));
typedef _Float16 half8  __attribute__((ext_vector_type(8)));
typedef _Float16 v16h   __attribute__((ext_vector_type(16)));
typedef float    v8f    __attribute__((ext_vector_type(8)));
typedef unsigned int u32x4 __attribute__((ext_vector_type(4)));
typedef int          i32x8 __attribute__((ext_vector_type(8)));

#define B_N   8192
#define D_K   64
#define INV_T 14.285714285714286f   /* 1 / 0.07 */
#define SHIFT INV_T                 /* fixed max-shift: s_ij <= 1/T */
#define K2    20.609929155556617f   /* INV_T * log2(e): exp(x)=exp2(x*log2e) */

#define CHUNK_ROWS  128
#define CHUNK_ELEMS (CHUNK_ROWS * D_K)   /* 8192 halfs  */
#define NCHUNK      (B_N / CHUNK_ROWS)   /* 64 chunks   */

// ---------------------------------------------------------------------------
// Kernel 1: L2-normalize rows (wave per row), write f16 An/Bn, f32 scaled diag
// ---------------------------------------------------------------------------
__global__ void k_normalize(const float* __restrict__ img,
                            const float* __restrict__ txt,
                            _Float16* __restrict__ An,
                            _Float16* __restrict__ Bn,
                            float*    __restrict__ diag)
{
    const int lane = threadIdx.x & 31;
    const int row  = blockIdx.x * (blockDim.x >> 5) + (threadIdx.x >> 5);
    if (row >= B_N) return;

    const float2 a = ((const float2*)(img + (size_t)row * D_K))[lane];
    const float2 b = ((const float2*)(txt + (size_t)row * D_K))[lane];

    float sa = a.x * a.x + a.y * a.y;
    float sb = b.x * b.x + b.y * b.y;
    float dd = a.x * b.x + a.y * b.y;
    #pragma unroll
    for (int m = 1; m < 32; m <<= 1) {
        sa += __shfl_xor(sa, m, 32);
        sb += __shfl_xor(sb, m, 32);
        dd += __shfl_xor(dd, m, 32);
    }
    const float ia = 1.0f / fmaxf(sqrtf(sa), 1e-12f);
    const float ib = 1.0f / fmaxf(sqrtf(sb), 1e-12f);

    half2v ah, bh;
    ah.x = (_Float16)(a.x * ia); ah.y = (_Float16)(a.y * ia);
    bh.x = (_Float16)(b.x * ib); bh.y = (_Float16)(b.y * ib);
    *(half2v*)(An + (size_t)row * D_K + lane * 2) = ah;
    *(half2v*)(Bn + (size_t)row * D_K + lane * 2) = bh;

    if (lane == 0) diag[row] = dd * ia * ib * INV_T;   // exact f32 diagonal
}

// ---------------------------------------------------------------------------
// 16-bit A/B fragment loader per CDNA5 ISA 16x32 layout:
//   lane l<16 : row r0+l,    K = {k0+0..7,  k0+16..23}
//   lane l>=16: row r0+l-16, K = {k0+8..15, k0+24..31}
// Two contiguous 16-byte loads per lane (global_load_b128 / ds_load_b128).
// ---------------------------------------------------------------------------
__device__ __forceinline__ v16h load_frag16(const _Float16* mat,
                                            int r0, int k0, int lane)
{
    const int l = lane & 15;
    const int h = (lane >> 4) & 1;
    const _Float16* p = mat + (size_t)(r0 + l) * D_K + k0 + h * 8;
    const half8 lo = *(const half8*)(p);        // K: k0 + 8h .. +7
    const half8 hi = *(const half8*)(p + 16);   // K: k0 + 16 + 8h .. +7
    v16h f;
    #pragma unroll
    for (int i = 0; i < 8; ++i) { f[i] = lo[i]; f[8 + i] = hi[i]; }
    return f;
}

// ---------------------------------------------------------------------------
// TDM: DMA one contiguous 16 KB chunk (128 B-rows) Global -> LDS.
// D# packed per CDNA5 ISA 8.3/8.4: count=1, type=2(image), data_size=2B,
// 1-D tile of CHUNK_ELEMS elements, huge tensor_dim0 (no OOB clipping).
// Issued via inline asm (portable across toolchains); VADDR2/VADDR3 = NULL
// (<=2-D tensor -> 2 SGPR-group form).  g0 -> s[n:n+3], g1 -> s[m:m+7].
// ---------------------------------------------------------------------------
__device__ __forceinline__ void tdm_load_chunk(const _Float16* gsrc,
                                               unsigned lds_off)
{
    const unsigned long long ga = (unsigned long long)(uintptr_t)gsrc;
    u32x4 g0;
    g0[0] = 1u;                                      // count=1, user load
    g0[1] = lds_off;                                 // lds_addr[31:0]
    g0[2] = (unsigned)(ga & 0xFFFFFFFFu);            // global_addr[31:0]
    g0[3] = (unsigned)((ga >> 32) & 0x1FFFFFFu)      // global_addr[56:32]
          | (2u << 30);                              // type = 2 (image)
    i32x8 g1;
    g1[0] = 0x00010000;                              // data_size=1 (2 bytes)
    g1[1] = 0;                                       // tensor_dim0 lo16 (0x80000)
    g1[2] = (int)((0x80000u >> 16) | (1u << 16));    // tensor_dim0 hi | tensor_dim1=1
    g1[3] = (int)(CHUNK_ELEMS << 16);                // tile_dim0 = 8192
    g1[4] = 1;                                       // tile_dim1 = 1, tile_dim2 = 0
    g1[5] = (int)CHUNK_ELEMS;                        // tensor_dim0_stride lo
    g1[6] = 0;
    g1[7] = 0;
    asm volatile("tensor_load_to_lds %0, %1"
                 :: "s"(g0), "s"(g1) : "memory");
}

__device__ __forceinline__ void wait_tensorcnt0()
{
#if __has_builtin(__builtin_amdgcn_s_wait_tensorcnt)
    __builtin_amdgcn_s_wait_tensorcnt(0);
#else
    asm volatile("s_wait_tensorcnt 0x0" ::: "memory");
#endif
}
__device__ __forceinline__ void wait_tensorcnt1()
{
#if __has_builtin(__builtin_amdgcn_s_wait_tensorcnt)
    __builtin_amdgcn_s_wait_tensorcnt(1);
#else
    asm volatile("s_wait_tensorcnt 0x1" ::: "memory");
#endif
}

// ---------------------------------------------------------------------------
// Kernel 2: block-cooperative exp-sum sweep.
//   blocks [0,64)  : rows of  S = An Bn^T   -> rowsum
//   blocks [64,128): rows of S^T = Bn An^T  -> colsum
// Each block owns 128 A-rows (16 per wave, fragments pinned in VGPRs) and
// sweeps all of B in 64 double-buffered LDS chunks staged by the TDM.
// ---------------------------------------------------------------------------
__global__ void __launch_bounds__(256)
k_expsums(const _Float16* __restrict__ An,
          const _Float16* __restrict__ Bn,
          float* __restrict__ rowsum,
          float* __restrict__ colsum)
{
    __shared__ _Float16 ldsB[2][CHUNK_ELEMS];   // 2 x 16 KB

    const int lane = threadIdx.x & 31;
    const int wave = threadIdx.x >> 5;

    const bool colside = blockIdx.x >= (B_N / CHUNK_ROWS);
    const int  bt      = colside ? (blockIdx.x - B_N / CHUNK_ROWS) : blockIdx.x;
    const _Float16* __restrict__ A  = colside ? Bn : An;
    const _Float16* __restrict__ Bm = colside ? An : Bn;
    float* __restrict__ out         = colside ? colsum : rowsum;
    const int r0 = bt * CHUNK_ROWS + wave * 16;

    // A fragments (16 rows x K=64) pinned in registers for the whole sweep.
    const v16h a0 = load_frag16(A, r0, 0,  lane);
    const v16h a1 = load_frag16(A, r0, 32, lane);

    const unsigned lds0 = (unsigned)(uintptr_t)&ldsB[0][0];
    const unsigned lds1 = (unsigned)(uintptr_t)&ldsB[1][0];

    if (wave == 0) tdm_load_chunk(Bm, lds0);     // preload chunk 0

    float acc[8];
    #pragma unroll
    for (int v = 0; v < 8; ++v) acc[v] = 0.0f;

    for (int ch = 0; ch < NCHUNK; ++ch) {
        if (wave == 0) {
            if (ch + 1 < NCHUNK) {   // issue next chunk into the other buffer
                tdm_load_chunk(Bm + (size_t)(ch + 1) * CHUNK_ELEMS,
                               (ch & 1) ? lds0 : lds1);
                wait_tensorcnt1();   // in-order TDM: chunk ch now complete
            } else {
                wait_tensorcnt0();
            }
        }
        __syncthreads();             // chunk ch visible to all waves

        const _Float16* bbase = &ldsB[ch & 1][0];
        #pragma unroll
        for (int cl = 0; cl < CHUNK_ROWS; cl += 16) {
            const v16h b0 = load_frag16(bbase, cl, 0,  lane);  // ds_load_b128
            const v16h b1 = load_frag16(bbase, cl, 32, lane);
            v8f c = {};
            c = __builtin_amdgcn_wmma_f32_16x16x32_f16(false, a0, false, b0,
                                                       (short)0, c, false, false);
            c = __builtin_amdgcn_wmma_f32_16x16x32_f16(false, a1, false, b1,
                                                       (short)0, c, false, false);
            // C layout: VGPR v, lanes 0..15 -> (M=v, N=lane),
            //                    lanes 16..31 -> (M=v+8, N=lane-16)
            // exp(s - SHIFT) = exp2((c - 1) * INV_T * log2e) -> one v_exp each
            #pragma unroll
            for (int v = 0; v < 8; ++v)
                acc[v] += exp2f((c[v] - 1.0f) * K2);
        }
        __syncthreads();             // protect buffer before it is re-filled
    }

    // Sum over N: reduce across the 16 lanes of each half-wave.
    #pragma unroll
    for (int v = 0; v < 8; ++v) {
        #pragma unroll
        for (int m = 1; m < 16; m <<= 1)
            acc[v] += __shfl_xor(acc[v], m, 16);
    }

    // lane 0 holds rows r0+0..7, lane 16 holds rows r0+8..15
    if ((lane & 15) == 0) {
        const int mo = (lane >> 4) * 8;
        #pragma unroll
        for (int v = 0; v < 8; ++v)
            out[r0 + mo + v] = acc[v];
    }
}

// ---------------------------------------------------------------------------
// Kernel 3: scalar finalize.  logZ = log(sum) + SHIFT;
// w = (1 - exp(dlr))^2 ; loss = sum_i w*(-dlr - dlc) / (2B)
// ---------------------------------------------------------------------------
__global__ void k_finalize(const float* __restrict__ diag,
                           const float* __restrict__ rowsum,
                           const float* __restrict__ colsum,
                           float* __restrict__ outp)
{
    __shared__ float red[1024];
    float part = 0.0f;
    for (int i = threadIdx.x; i < B_N; i += blockDim.x) {
        const float d   = diag[i];
        const float dlr = d - (logf(rowsum[i]) + SHIFT);
        const float dlc = d - (logf(colsum[i]) + SHIFT);
        const float p   = expf(dlr);
        const float om  = 1.0f - p;
        const float w   = om * om;              // ALPHA=1, GAMMA=2
        part += w * (-dlr - dlc);
    }
    red[threadIdx.x] = part;
    __syncthreads();
    for (int s = blockDim.x >> 1; s > 0; s >>= 1) {
        if (threadIdx.x < (unsigned)s) red[threadIdx.x] += red[threadIdx.x + s];
        __syncthreads();
    }
    if (threadIdx.x == 0) outp[0] = red[0] / (2.0f * (float)B_N);
}

// ---------------------------------------------------------------------------
extern "C" void kernel_launch(void* const* d_in, const int* in_sizes, int n_in,
                              void* d_out, int out_size, void* d_ws, size_t ws_size,
                              hipStream_t stream)
{
    (void)in_sizes; (void)n_in; (void)out_size; (void)ws_size;
    const float* img = (const float*)d_in[0];   // [8192,64] f32
    const float* txt = (const float*)d_in[1];   // [8192,64] f32
    float* outp = (float*)d_out;                // scalar f32

    char* ws = (char*)d_ws;
    _Float16* An   = (_Float16*)ws;                                 // 1 MB
    _Float16* Bn   = (_Float16*)(ws + (size_t)B_N * D_K * 2);       // 1 MB
    float*    diag = (float*)(ws + (size_t)B_N * D_K * 4);          // 32 KB
    float*    rsum = diag + B_N;                                    // 32 KB
    float*    csum = rsum + B_N;                                    // 32 KB

    // 1 wave per row, 8 waves per block
    k_normalize<<<B_N / 8, 256, 0, stream>>>(img, txt, An, Bn, diag);

    // 128 blocks: 64 row-side + 64 col-side, 8 waves x 16 rows each
    k_expsums<<<2 * (B_N / CHUNK_ROWS), 256, 0, stream>>>(An, Bn, rsum, csum);

    k_finalize<<<1, 1024, 0, stream>>>(diag, rsum, csum, outp);
}
